// NodeTaskHead_6949257085369
// MI455X (gfx1250) — matrix operand order repaired and checked by reading, not compiled, not run on hardware
//
#include <hip/hip_runtime.h>
#include <math.h>

#define BB 4
#define NN 512
#define EE 768
#define HH 12
#define DD 64
#define BN (BB * NN)

typedef __attribute__((ext_vector_type(16))) _Float16 v16h;
typedef __attribute__((ext_vector_type(8)))  _Float16 v8h;
typedef __attribute__((ext_vector_type(8)))  float    v8f;

// ---------------- WMMA fragment helpers (CDNA5 16x16x32 f16 layouts) ----------------

// A-matrix 16x32 (MxK), row-major source, lda in halves.
// lane l<16: M=l,   halves 0..7 -> K=0..7,  halves 8..15 -> K=16..23
// lane l>=16: M=l-16, halves 0..7 -> K=8..15, halves 8..15 -> K=24..31
__device__ __forceinline__ v16h load_a_frag(const _Float16* base, int lda, int kb, int lane) {
  int l16 = lane & 15;
  int sel = (lane >> 4) << 3;              // 0 or 8
  const _Float16* p = base + (size_t)l16 * lda + kb + sel;
  v8h lo = *(const v8h*)(p);               // 16B aligned
  v8h hi = *(const v8h*)(p + 16);
  v16h r;
#pragma unroll
  for (int i = 0; i < 8; ++i) { r[i] = lo[i]; r[i + 8] = hi[i]; }
  return r;
}

// B-matrix 32x16 (KxN) from an N-major (transposed) f16 array: row n holds K contiguous.
// lane l<16: N=l, halves 0..7 -> K=0..7, halves 8..15 -> K=16..23 ; lane l>=16: N=l-16, K=8..15, 24..31
__device__ __forceinline__ v16h load_b_frag(const _Float16* base, int ldb, int kb, int lane) {
  int l16 = lane & 15;
  int sel = (lane >> 4) << 3;              // 0 or 8
  const _Float16* p = base + (size_t)l16 * ldb + kb + sel;
  v8h lo = *(const v8h*)(p);
  v8h hi = *(const v8h*)(p + 16);
  v16h r;
#pragma unroll
  for (int i = 0; i < 8; ++i) { r[i] = lo[i]; r[i + 8] = hi[i]; }
  return r;
}

__device__ __forceinline__ v8f wmma_f16(v16h a, v16h b, v8f c) {
  return __builtin_amdgcn_wmma_f32_16x16x32_f16(false, a, false, b, (short)0, c, false, false);
}

// C/D 16x16 f32: VGPR g, lane l -> (M = 8*(l>=16)+g, N = l&15)
__device__ __forceinline__ void store_c_f16(_Float16* out, int ldo, int lane, v8f c, float scale) {
  int l16 = lane & 15;
  int mb = (lane >> 4) << 3;
#pragma unroll
  for (int g = 0; g < 8; ++g)
    out[(size_t)(mb + g) * ldo + l16] = (_Float16)(c[g] * scale);
}

__device__ __forceinline__ void store_c_lds(float* sm, int ld, int lane, v8f c) {
  int l16 = lane & 15;
  int mb = (lane >> 4) << 3;
#pragma unroll
  for (int g = 0; g < 8; ++g)
    sm[(mb + g) * ld + l16] = c[g];
}

// ---------------- prep: f32 -> f16 copies ----------------

__global__ void cvt_f16_kernel(const float* __restrict__ x, _Float16* __restrict__ y, int n) {
  int i = blockIdx.x * blockDim.x + threadIdx.x;
  if (i < n) y[i] = (_Float16)x[i];
}

// Wt[n*E + k] = (f16) W[k*E + n], coalesced via padded LDS tile.
__global__ void __launch_bounds__(256) transpose_w_kernel(const float* __restrict__ W,
                                                          _Float16* __restrict__ Wt) {
  __shared__ _Float16 tile[16][17];
  int tx = threadIdx.x & 15;
  int ty = threadIdx.x >> 4;
  int k0 = blockIdx.x * 16;
  int n0 = blockIdx.y * 16;
  // coalesced read: row k0+ty, cols n0+tx
  tile[ty][tx] = (_Float16)W[(size_t)(k0 + ty) * EE + n0 + tx];
  __syncthreads();
  // coalesced write: Wt row n0+ty, cols k0+tx
  Wt[(size_t)(n0 + ty) * EE + k0 + tx] = tile[tx][ty];
}

// ---------------- QKV projection GEMM: [2048x768] @ [768x768] x3, f16 WMMA ----------------
// block = 128 threads (4 waves). Each wave computes a 32x64 output tile (2x4 WMMA tiles),
// K double-buffered in registers (unroll 64) so loads overlap the matrix pipe.
// grid.y picks the 256-col band, grid.z picks {q,k,v}. q scaled by D^-0.5 in the epilogue.
__global__ void __launch_bounds__(128) gemm_qkv_kernel(const _Float16* __restrict__ A,
                                                       const _Float16* __restrict__ Wt3,
                                                       _Float16* __restrict__ out3) {
  int lane = threadIdx.x & 31;
  int wv   = threadIdx.x >> 5;                    // 0..3
  int m0   = blockIdx.x * 32;
  int n0   = blockIdx.y * 256 + wv * 64;
  int z    = blockIdx.z;
  const _Float16* Wt  = Wt3  + (size_t)z * EE * EE;
  _Float16*       out = out3 + (size_t)z * BN * EE;
  float scale = (z == 0) ? 0.125f : 1.0f;         // D^-0.5 folded into q

  const _Float16* arow[2];
  const _Float16* brow[4];
#pragma unroll
  for (int mi = 0; mi < 2; ++mi) arow[mi] = A + (size_t)(m0 + mi * 16) * EE;
#pragma unroll
  for (int ni = 0; ni < 4; ++ni) brow[ni] = Wt + (size_t)(n0 + ni * 16) * EE;

  v8f zero = {};
  v8f acc[2][4];
#pragma unroll
  for (int mi = 0; mi < 2; ++mi)
#pragma unroll
    for (int ni = 0; ni < 4; ++ni) acc[mi][ni] = zero;

  v16h a0[2], a1[2], b0[4], b1[4];
#pragma unroll
  for (int mi = 0; mi < 2; ++mi) a0[mi] = load_a_frag(arow[mi], EE, 0, lane);
#pragma unroll
  for (int ni = 0; ni < 4; ++ni) b0[ni] = load_b_frag(brow[ni], EE, 0, lane);

  for (int kb = 0; kb < EE; kb += 64) {
    // prefetch k-step kb+32 into buffer 1
#pragma unroll
    for (int mi = 0; mi < 2; ++mi) a1[mi] = load_a_frag(arow[mi], EE, kb + 32, lane);
#pragma unroll
    for (int ni = 0; ni < 4; ++ni) b1[ni] = load_b_frag(brow[ni], EE, kb + 32, lane);
    // compute on buffer 0 (loads above still in flight)
#pragma unroll
    for (int mi = 0; mi < 2; ++mi)
#pragma unroll
      for (int ni = 0; ni < 4; ++ni) acc[mi][ni] = wmma_f16(a0[mi], b0[ni], acc[mi][ni]);
    // prefetch k-step kb+64 into buffer 0 (wrapped dummy load on last iter)
    int kn = (kb + 64 < EE) ? (kb + 64) : 0;
#pragma unroll
    for (int mi = 0; mi < 2; ++mi) a0[mi] = load_a_frag(arow[mi], EE, kn, lane);
#pragma unroll
    for (int ni = 0; ni < 4; ++ni) b0[ni] = load_b_frag(brow[ni], EE, kn, lane);
    // compute on buffer 1
#pragma unroll
    for (int mi = 0; mi < 2; ++mi)
#pragma unroll
      for (int ni = 0; ni < 4; ++ni) acc[mi][ni] = wmma_f16(a1[mi], b1[ni], acc[mi][ni]);
  }

#pragma unroll
  for (int mi = 0; mi < 2; ++mi)
#pragma unroll
    for (int ni = 0; ni < 4; ++ni)
      store_c_f16(out + (size_t)(m0 + mi * 16) * EE + n0 + ni * 16, EE, lane,
                  acc[mi][ni], scale);
}

// ---------------- s[b,h,j] = sum_d v[b,h,j,d] * Wf[h*D+d] ----------------
__global__ void vs_kernel(const _Float16* __restrict__ vh, const float* __restrict__ Wf,
                          float* __restrict__ s) {
  int idx = blockIdx.x * blockDim.x + threadIdx.x;
  if (idx >= BB * HH * NN) return;
  int j = idx % NN;
  int h = (idx / NN) % HH;
  int b = idx / (NN * HH);
  const _Float16* vp = vh + ((size_t)(b * NN + j)) * EE + h * DD;
  const float*    wf = Wf + h * DD;
  float acc = 0.f;
#pragma unroll
  for (int d = 0; d < DD; ++d) acc += (float)vp[d] * wf[d];
  s[idx] = acc;                                   // layout [b,h,j]
}

// ---------------- fused attention: scores (WMMA) -> softmax -> w[b,i,j] = sum_h p*s ------
// block = 256 threads (8 waves). Block owns 16 rows (i) x 512 cols (j) for one b; loops h.
// All 10 fragment loads for the 16x64 score band are issued up front; the 8 WMMAs then run
// back-to-back. Softmax threads own (row = tid>>4, 32-col chunk) and accumulate w across
// heads in registers, writing w once.
__global__ void __launch_bounds__(256) attn_kernel(const _Float16* __restrict__ qh,
                                                   const _Float16* __restrict__ kh,
                                                   const float* __restrict__ s,
                                                   const unsigned char* __restrict__ mask,
                                                   float* __restrict__ w) {
  __shared__ float Ssc[16 * NN];                  // 32 KB score tile
  __shared__ float red[16 * 16];
  __shared__ float rmax[16];
  __shared__ float rsum[16];

  int tid  = threadIdx.x;
  int lane = tid & 31, wv = tid >> 5;
  int i0 = blockIdx.x * 16;
  int b  = blockIdx.y;

  int r   = tid >> 4;                             // owned row 0..15
  int c16 = tid & 15;
  int c0  = c16 * 32;                             // owned 32-col chunk

  unsigned mbits = 0u;
  const unsigned char* mrow = mask + ((size_t)b * NN + (i0 + r)) * NN + c0;
#pragma unroll
  for (int t = 0; t < 32; ++t) mbits |= (unsigned)(mrow[t] != 0) << t;

  float wreg[32];
#pragma unroll
  for (int t = 0; t < 32; ++t) wreg[t] = 0.f;

  for (int h = 0; h < HH; ++h) {
    // scores: S[i,j] = sum_d q[i,d] * k[j,d]; K = D = 64 -> two k-steps, fully unrolled
    const _Float16* qb  = qh + ((size_t)b * NN + i0) * EE + h * DD;
    const _Float16* kbp = kh + ((size_t)b * NN) * EE + h * DD;
    int j0 = wv * 64;

    v16h a0 = load_a_frag(qb, EE, 0, lane);
    v16h a1 = load_a_frag(qb, EE, 32, lane);
    v16h bf0[4], bf1[4];
#pragma unroll
    for (int sb = 0; sb < 4; ++sb)
      bf0[sb] = load_b_frag(kbp + (size_t)(j0 + sb * 16) * EE, EE, 0, lane);
#pragma unroll
    for (int sb = 0; sb < 4; ++sb)
      bf1[sb] = load_b_frag(kbp + (size_t)(j0 + sb * 16) * EE, EE, 32, lane);

    v8f zero = {};
    v8f acc[4] = {zero, zero, zero, zero};
#pragma unroll
    for (int sb = 0; sb < 4; ++sb) acc[sb] = wmma_f16(a0, bf0[sb], acc[sb]);
#pragma unroll
    for (int sb = 0; sb < 4; ++sb) acc[sb] = wmma_f16(a1, bf1[sb], acc[sb]);

#pragma unroll
    for (int sb = 0; sb < 4; ++sb)
      store_c_lds(&Ssc[j0 + sb * 16], NN, lane, acc[sb]);
    __syncthreads();

    // row max
    float lmax = -3.0e38f;
#pragma unroll
    for (int t = 0; t < 32; ++t) {
      float v = Ssc[r * NN + c0 + t];
      if ((mbits >> t) & 1) v = -3.0e38f;
      lmax = fmaxf(lmax, v);
    }
    red[r * 16 + c16] = lmax;
    __syncthreads();
    if (c16 == 0) {
      float m = red[r * 16];
#pragma unroll
      for (int t = 1; t < 16; ++t) m = fmaxf(m, red[r * 16 + t]);
      rmax[r] = m;
    }
    __syncthreads();

    // exp + row sum
    float rm = rmax[r];
    float pv[32];
    float lsum = 0.f;
#pragma unroll
    for (int t = 0; t < 32; ++t) {
      float v = Ssc[r * NN + c0 + t];
      float e = ((mbits >> t) & 1) ? 0.f : __expf(v - rm);
      pv[t] = e;
      lsum += e;
    }
    red[r * 16 + c16] = lsum;
    __syncthreads();
    if (c16 == 0) {
      float m = 0.f;
#pragma unroll
      for (int t = 0; t < 16; ++t) m += red[r * 16 + t];
      rsum[r] = m;
    }
    __syncthreads();

    // w += p * s[b,h,:]
    float inv = 1.0f / rsum[r];
    const float* sp = s + ((size_t)b * HH + h) * NN + c0;
#pragma unroll
    for (int t = 0; t < 32; ++t) wreg[t] += pv[t] * inv * sp[t];
    __syncthreads();                               // before next h overwrites Ssc
  }

  float* wp = w + ((size_t)b * NN + (i0 + r)) * NN + c0;
#pragma unroll
  for (int t = 0; t < 32; ++t) wp[t] = wreg[t];
}

// ---------------- force[b,i,c] = sum_j w[b,i,j] * dp[b,i,j,c] ----------------
__global__ void __launch_bounds__(128) force_kernel(const float* __restrict__ w,
                                                    const float* __restrict__ dp,
                                                    float* __restrict__ out) {
  int tid = threadIdx.x;
  int bi  = blockIdx.x;                            // (b*N + i)
  const float* wp  = w  + (size_t)bi * NN;
  const float* dpp = dp + (size_t)bi * NN * 3;
  float a0 = 0.f, a1 = 0.f, a2 = 0.f;
  for (int j = tid; j < NN; j += 128) {
    float ww = wp[j];                              // already 0 at masked positions
    a0 += ww * dpp[j * 3 + 0];
    a1 += ww * dpp[j * 3 + 1];
    a2 += ww * dpp[j * 3 + 2];
  }
  __shared__ float red[3][128];
  red[0][tid] = a0; red[1][tid] = a1; red[2][tid] = a2;
  __syncthreads();
  for (int st = 64; st > 0; st >>= 1) {
    if (tid < st) {
      red[0][tid] += red[0][tid + st];
      red[1][tid] += red[1][tid + st];
      red[2][tid] += red[2][tid + st];
    }
    __syncthreads();
  }
  if (tid == 0) {
    out[(size_t)bi * 3 + 0] = red[0][0];
    out[(size_t)bi * 3 + 1] = red[1][0];
    out[(size_t)bi * 3 + 2] = red[2][0];
  }
}

// ---------------- launcher ----------------
extern "C" void kernel_launch(void* const* d_in, const int* in_sizes, int n_in,
                              void* d_out, int out_size, void* d_ws, size_t ws_size,
                              hipStream_t stream) {
  (void)in_sizes; (void)n_in; (void)out_size; (void)ws_size;
  const float*         query = (const float*)d_in[0];
  const float*         dp    = (const float*)d_in[1];
  const unsigned char* mask  = (const unsigned char*)d_in[2];
  const float*         Wq    = (const float*)d_in[3];
  const float*         Wk    = (const float*)d_in[4];
  const float*         Wv    = (const float*)d_in[5];
  const float*         Wf    = (const float*)d_in[6];
  float* out = (float*)d_out;

  char* ws = (char*)d_ws;
  size_t off = 0;
  auto salloc = [&](size_t bytes) {
    void* p = ws + off;
    off = (off + bytes + 255) & ~(size_t)255;
    return p;
  };
  _Float16* qin  = (_Float16*)salloc((size_t)BN * EE * 2);       // f16 query
  _Float16* Wt   = (_Float16*)salloc((size_t)3 * EE * EE * 2);   // N-major f16 weights
  _Float16* qkv  = (_Float16*)salloc((size_t)3 * BN * EE * 2);   // f16 q,k,v ([b,i,h,d])
  float*    svec = (float*)salloc((size_t)BB * HH * NN * 4);     // s[b,h,j]
  float*    wmat = (float*)salloc((size_t)BB * NN * NN * 4);     // w[b,i,j]

  cvt_f16_kernel<<<(BN * EE + 255) / 256, 256, 0, stream>>>(query, qin, BN * EE);
  transpose_w_kernel<<<dim3(EE / 16, EE / 16), 256, 0, stream>>>(Wq, Wt);
  transpose_w_kernel<<<dim3(EE / 16, EE / 16), 256, 0, stream>>>(Wk, Wt + (size_t)EE * EE);
  transpose_w_kernel<<<dim3(EE / 16, EE / 16), 256, 0, stream>>>(Wv, Wt + (size_t)2 * EE * EE);

  gemm_qkv_kernel<<<dim3(BN / 32, 3, 3), 128, 0, stream>>>(qin, Wt, qkv);

  vs_kernel<<<(BB * HH * NN + 255) / 256, 256, 0, stream>>>(qkv + (size_t)2 * BN * EE, Wf, svec);

  attn_kernel<<<dim3(NN / 16, BB), 256, 0, stream>>>(qkv, qkv + (size_t)BN * EE,
                                                     svec, mask, wmat);

  force_kernel<<<BN, 128, 0, stream>>>(wmat, dp, out);
}
